// HyperbolicNN_21268678050357
// MI455X (gfx1250) — matrix operand order repaired and checked by reading, not compiled, not run on hardware
//
#include <hip/hip_runtime.h>
#include <hip/hip_bf16.h>

typedef __attribute__((ext_vector_type(16))) __bf16 bf16x16;
typedef __attribute__((ext_vector_type(8)))  __bf16 bf16x8;
typedef __attribute__((ext_vector_type(8)))  float  floatx8;
typedef __attribute__((ext_vector_type(4)))  float  floatx4;
typedef __attribute__((ext_vector_type(4)))  unsigned int uint32x4;
typedef __attribute__((ext_vector_type(8)))  int  int32x8;
typedef __attribute__((ext_vector_type(4)))  int  int32x4;

#define MIN_NORM 1e-15f
#define MAX_RAD  0.996f          /* 1 - 4e-3 (geoopt float32 proj eps) */
#define ATANH_CL (1.0f - 1e-7f)

// sched_group_barrier masks
#define SGB_WMMA 0x008
#define SGB_DSRD 0x100

__device__ __forceinline__ float wred32(float v) {
#pragma unroll
    for (int m = 16; m > 0; m >>= 1) v += __shfl_xor(v, m, 32);
    return v;
}

__device__ __forceinline__ float artanh_pos(float z) {
    z = fminf(z, ATANH_CL);
    return 0.5f * logf((1.0f + z) / (1.0f - z));
}

template <int NV>
__device__ __forceinline__ float proj_row(float* v) {
    float s = 0.f;
#pragma unroll
    for (int j = 0; j < NV; ++j) s += v[j] * v[j];
    float n = fmaxf(sqrtf(wred32(s)), MIN_NORM);
    if (n > MAX_RAD) {
        float f = MAX_RAD / n;
#pragma unroll
        for (int j = 0; j < NV; ++j) v[j] *= f;
        n = MAX_RAD;
    }
    return n;
}

__device__ __forceinline__ bf16x16 load_bfrag(const __bf16* p) {
    bf16x8 lo = *(const bf16x8*)p;
    bf16x8 hi = *(const bf16x8*)(p + 16);
    return __builtin_shufflevector(lo, hi, 0, 1, 2, 3, 4, 5, 6, 7,
                                   8, 9, 10, 11, 12, 13, 14, 15);
}

__device__ __forceinline__ bf16x16 load_afrag_lds(const __bf16* rowp, int kt,
                                                  int aklo, int akhi) {
    bf16x8 lo = *(const bf16x8*)(rowp + kt * 32 + aklo);
    bf16x8 hi = *(const bf16x8*)(rowp + kt * 32 + akhi);
    return __builtin_shufflevector(lo, hi, 0, 1, 2, 3, 4, 5, 6, 7,
                                   8, 9, 10, 11, 12, 13, 14, 15);
}

// ---- Tensor Data Mover: 2D bf16 tile (tile_d1 rows x tile_d0 elems) -> LDS ----
__device__ __forceinline__ void tdm_load_2d(const __bf16* gsrc, unsigned lds_byte_addr,
                                            unsigned tensor_d0, unsigned tensor_d1,
                                            unsigned tile_d0, unsigned tile_d1,
                                            unsigned stride0) {
    unsigned long long ga = (unsigned long long)(uintptr_t)gsrc;
    uint32x4 g0;
    g0[0] = 1u;                                                 // count=1, user desc
    g0[1] = lds_byte_addr;                                      // lds_addr
    g0[2] = (unsigned)ga;                                       // global_addr[31:0]
    g0[3] = (unsigned)((ga >> 32) & 0x1FFFFFFu) | (2u << 30);   // [56:32] | type=2
    int32x8 g1;
    g1[0] = (int)(1u << 16);                                    // data_size=1 (2 bytes)
    g1[1] = (int)((tensor_d0 & 0xFFFFu) << 16);                 // tensor_dim0[15:0]
    g1[2] = (int)(((tensor_d0 >> 16) & 0xFFFFu) | ((tensor_d1 & 0xFFFFu) << 16));
    g1[3] = (int)(((tensor_d1 >> 16) & 0xFFFFu) | ((tile_d0 & 0xFFFFu) << 16));
    g1[4] = (int)(tile_d1 & 0xFFFFu);                           // tile_dim1 (tile_dim2=0)
    g1[5] = (int)stride0;                                       // tensor_dim0_stride[31:0]
    g1[6] = 0;
    g1[7] = 0;
    int32x4 z4 = {0, 0, 0, 0};
    int32x8 z8 = {0, 0, 0, 0, 0, 0, 0, 0};
    __builtin_amdgcn_tensor_load_to_lds(g0, g1, z4, z4, z8, 0);
}

// one-shot weight fp32 -> bf16 conversion into workspace
__global__ void hypnn_cvt_w(const float* __restrict__ w1, const float* __restrict__ w2,
                            __bf16* __restrict__ o1, __bf16* __restrict__ o2) {
    int i = blockIdx.x * blockDim.x + threadIdx.x;   // 131072 threads
    o1[i] = (__bf16)w1[i];                           // 512*256
    if (i < 128 * 512) o2[i] = (__bf16)w2[i];
}

__global__ __launch_bounds__(128, 1)
void hypnn_fused(const float* __restrict__ x,  const __bf16* __restrict__ w1b,
                 const float* __restrict__ b1, const __bf16* __restrict__ w2b,
                 const float* __restrict__ b2, float* __restrict__ out) {
    __shared__ __align__(16) __bf16 s_h[64 * 512];    // 64 KB: h then h2 (bf16)
    __shared__ __align__(16) float  s_mx2[64 * 128];  // 32 KB: layer-2 matvec result
    __shared__ __align__(16) __bf16 s_stage[16384];   // 32 KB: TDM weight staging (2 bufs)
    __shared__ float s_xn1[64];                       // ||x_row||^2
    __shared__ float s_xn2[64];                       // ||h2_row|| feeding layer 2

    const int lane  = threadIdx.x & 31;
    const int wave  = threadIdx.x >> 5;
    const int l15   = lane & 15;
    const int lhalf = (lane >= 16);
    const int rowBase = blockIdx.x * 64;
    const unsigned stageLds = (unsigned)(uintptr_t)(&s_stage[0]);  // LDS byte addr

    // ---- bias fragments (BLOCKED partition: lane owns cols lane*V..lane*V+V-1) ----
    float bv1[16], bv2[4];
    float yb = 0.f;
#pragma unroll
    for (int q = 0; q < 4; ++q) {
        floatx4 t = *(const floatx4*)(b1 + lane * 16 + q * 4);
#pragma unroll
        for (int k = 0; k < 4; ++k) { bv1[q * 4 + k] = t[k]; yb += t[k] * t[k]; }
    }
    const float y2b1 = wred32(yb);
    float yc = 0.f;
    {
        floatx4 t = *(const floatx4*)(b2 + lane * 4);
#pragma unroll
        for (int k = 0; k < 4; ++k) { bv2[k] = t[k]; yc += t[k] * t[k]; }
    }
    const float y2b2 = wred32(yc);

    // ================= A-fragments of x (+ row norms for free) =================
    bf16x16 afrag[8];
    {
        const float* xrow = x + (size_t)(rowBase + wave * 16 + l15) * 256;
        const int klo = lhalf ? 8  : 0;
        const int khi = lhalf ? 24 : 16;
        float axs = 0.f;
#pragma unroll
        for (int kt = 0; kt < 8; ++kt) {
            floatx8 lo = *(const floatx8*)(xrow + kt * 32 + klo);
            floatx8 hi = *(const floatx8*)(xrow + kt * 32 + khi);
            bf16x16 a;
#pragma unroll
            for (int i = 0; i < 8; ++i) {
                axs += lo[i] * lo[i] + hi[i] * hi[i];
                a[i] = (__bf16)lo[i]; a[8 + i] = (__bf16)hi[i];
            }
            afrag[kt] = a;
        }
        axs += __shfl_xor(axs, 16, 32);     // lanes l / l+16 hold complementary halves
        if (lane < 16) s_xn1[wave * 16 + l15] = axs;
    }

    // ================= GEMM1: h = x @ W1^T, W1 tiles DMA'd by TDM =================
    const int bko = lhalf ? 8 : 0;
    if (wave == 0)
        tdm_load_2d(w1b, stageLds, 256, 512, 256, 16, 256);
    for (int nt = 0; nt < 32; ++nt) {
        if (wave == 0) {
            if (nt < 31) {
                tdm_load_2d(w1b + (size_t)(nt + 1) * 16 * 256,
                            stageLds + ((nt + 1) & 1) * 8192u, 256, 512, 256, 16, 256);
                __builtin_amdgcn_s_wait_tensorcnt(1);
            } else {
                __builtin_amdgcn_s_wait_tensorcnt(0);
            }
        }
        __syncthreads();
        const __bf16* brow = s_stage + (nt & 1) * 4096 + l15 * 256 + bko;
        bf16x16 bf[8];
#pragma unroll
        for (int kt = 0; kt < 8; ++kt) bf[kt] = load_bfrag(brow + kt * 32);
        floatx8 acc = {};
#pragma unroll
        for (int kt = 0; kt < 8; ++kt)
            acc = __builtin_amdgcn_wmma_f32_16x16x32_bf16(false, afrag[kt], false, bf[kt],
                                                          (short)0, acc, false, false);
        // pin pipeline: 4 DS reads ahead, then alternate (1 WMMA : 2 DS reads)
        __builtin_amdgcn_sched_group_barrier(SGB_DSRD, 4, 0);
#pragma unroll
        for (int g = 0; g < 6; ++g) {
            __builtin_amdgcn_sched_group_barrier(SGB_WMMA, 1, 0);
            __builtin_amdgcn_sched_group_barrier(SGB_DSRD, 2, 0);
        }
        __builtin_amdgcn_sched_group_barrier(SGB_WMMA, 2, 0);

        const int drow = wave * 16 + (lhalf ? 8 : 0);
        const int dcol = nt * 16 + l15;
#pragma unroll
        for (int r = 0; r < 8; ++r) s_h[(drow + r) * 512 + dcol] = (__bf16)acc[r];
        __syncthreads();
    }

    // ================= Layer-1 Mobius pipeline (wave per row, blocked 16 cols/lane) ====
    for (int i = 0; i < 16; ++i) {
        const int rl = wave * 16 + i;
        float hv[16];
        {
            bf16x8 p0 = *(const bf16x8*)(&s_h[rl * 512 + lane * 16]);
            bf16x8 p1 = *(const bf16x8*)(&s_h[rl * 512 + lane * 16 + 8]);
#pragma unroll
            for (int j = 0; j < 8; ++j) { hv[j] = (float)p0[j]; hv[8 + j] = (float)p1[j]; }
        }

        const float xn = fmaxf(sqrtf(s_xn1[rl]), MIN_NORM);

        float ms = 0.f;
#pragma unroll
        for (int j = 0; j < 16; ++j) ms += hv[j] * hv[j];
        const float mxn = fmaxf(sqrtf(wred32(ms)), MIN_NORM);
        const float sc  = tanhf(mxn / xn * artanh_pos(xn)) / mxn;
#pragma unroll
        for (int j = 0; j < 16; ++j) hv[j] *= sc;

        float n1 = proj_row<16>(hv);

        // mobius_add(h, b1)
        float xy = 0.f;
#pragma unroll
        for (int j = 0; j < 16; ++j) xy += hv[j] * bv1[j];
        xy = wred32(xy);
        const float x2  = n1 * n1;
        const float den = fmaxf(1.f + 2.f * xy + x2 * y2b1, MIN_NORM);
        const float ca  = (1.f + 2.f * xy + y2b1) / den;
        const float cb  = (1.f - x2) / den;
#pragma unroll
        for (int j = 0; j < 16; ++j) hv[j] = ca * hv[j] + cb * bv1[j];
        proj_row<16>(hv);

        // expmap0(tanh(logmap0(h)))
        float s2 = 0.f;
#pragma unroll
        for (int j = 0; j < 16; ++j) s2 += hv[j] * hv[j];
        const float n2 = fmaxf(sqrtf(wred32(s2)), MIN_NORM);
        const float a2 = artanh_pos(n2) / n2;
        float ts = 0.f;
#pragma unroll
        for (int j = 0; j < 16; ++j) { hv[j] = tanhf(hv[j] * a2); ts += hv[j] * hv[j]; }
        const float tn = fmaxf(sqrtf(wred32(ts)), MIN_NORM);
        const float ef = tanhf(tn) / tn;
#pragma unroll
        for (int j = 0; j < 16; ++j) hv[j] *= ef;
        const float fn = proj_row<16>(hv);

        if (lane == 0) s_xn2[rl] = fn;
        {
            bf16x8 q0, q1;
#pragma unroll
            for (int j = 0; j < 8; ++j) { q0[j] = (__bf16)hv[j]; q1[j] = (__bf16)hv[8 + j]; }
            *(bf16x8*)(&s_h[rl * 512 + lane * 16])     = q0;
            *(bf16x8*)(&s_h[rl * 512 + lane * 16 + 8]) = q1;
        }
    }
    __syncthreads();

    // ================= GEMM2: mx2 = h2 @ W2^T (A from LDS, W2 via TDM) =================
    {
        const __bf16* arow_p = s_h + (wave * 16 + l15) * 512;
        const int aklo = lhalf ? 8  : 0;
        const int akhi = lhalf ? 24 : 16;
        if (wave == 0)
            tdm_load_2d(w2b, stageLds, 512, 128, 512, 16, 512);
        for (int nt = 0; nt < 8; ++nt) {
            if (wave == 0) {
                if (nt < 7) {
                    tdm_load_2d(w2b + (size_t)(nt + 1) * 16 * 512,
                                stageLds + ((nt + 1) & 1) * 16384u, 512, 128, 512, 16, 512);
                    __builtin_amdgcn_s_wait_tensorcnt(1);
                } else {
                    __builtin_amdgcn_s_wait_tensorcnt(0);
                }
            }
            __syncthreads();
            const __bf16* brow = s_stage + (nt & 1) * 8192 + l15 * 512 + bko;
            // 2-deep chunked pipeline over K (4 chunks of 4 WMMAs)
            bf16x16 af[2][4], bfr[2][4];
#pragma unroll
            for (int u = 0; u < 4; ++u) {
                af[0][u]  = load_afrag_lds(arow_p, u, aklo, akhi);
                bfr[0][u] = load_bfrag(brow + u * 32);
            }
            floatx8 acc = {};
#pragma unroll
            for (int c = 0; c < 4; ++c) {
                if (c < 3) {
#pragma unroll
                    for (int u = 0; u < 4; ++u) {
                        const int kt = (c + 1) * 4 + u;
                        af[(c + 1) & 1][u]  = load_afrag_lds(arow_p, kt, aklo, akhi);
                        bfr[(c + 1) & 1][u] = load_bfrag(brow + kt * 32);
                    }
                }
#pragma unroll
                for (int u = 0; u < 4; ++u)
                    acc = __builtin_amdgcn_wmma_f32_16x16x32_bf16(false, af[c & 1][u],
                                                                  false, bfr[c & 1][u],
                                                                  (short)0, acc, false, false);
            }
            // pin pipeline: chunk0's 16 DS reads, then (1 WMMA : 4 DS reads) x12, tail WMMAs
            __builtin_amdgcn_sched_group_barrier(SGB_DSRD, 16, 0);
#pragma unroll
            for (int g = 0; g < 12; ++g) {
                __builtin_amdgcn_sched_group_barrier(SGB_WMMA, 1, 0);
                __builtin_amdgcn_sched_group_barrier(SGB_DSRD, 4, 0);
            }
            __builtin_amdgcn_sched_group_barrier(SGB_WMMA, 4, 0);

            const int drow = wave * 16 + (lhalf ? 8 : 0);
            const int dcol = nt * 16 + l15;
#pragma unroll
            for (int r = 0; r < 8; ++r) s_mx2[(drow + r) * 128 + dcol] = acc[r];
            __syncthreads();
        }
    }

    // ================= Layer-2 Mobius epilogue (blocked 4 cols/lane) -> out ==========
    for (int i = 0; i < 16; ++i) {
        const int rl  = wave * 16 + i;
        const int row = rowBase + rl;
        float mv[4];
        {
            floatx4 t = *(const floatx4*)(&s_mx2[rl * 128 + lane * 4]);
#pragma unroll
            for (int j = 0; j < 4; ++j) mv[j] = t[j];
        }

        float ms = 0.f;
#pragma unroll
        for (int j = 0; j < 4; ++j) ms += mv[j] * mv[j];
        const float mxn = fmaxf(sqrtf(wred32(ms)), MIN_NORM);
        const float xn  = fmaxf(s_xn2[rl], MIN_NORM);
        const float sc  = tanhf(mxn / xn * artanh_pos(xn)) / mxn;
#pragma unroll
        for (int j = 0; j < 4; ++j) mv[j] *= sc;

        float n1 = proj_row<4>(mv);

        float xy = 0.f;
#pragma unroll
        for (int j = 0; j < 4; ++j) xy += mv[j] * bv2[j];
        xy = wred32(xy);
        const float x2  = n1 * n1;
        const float den = fmaxf(1.f + 2.f * xy + x2 * y2b2, MIN_NORM);
        const float ca  = (1.f + 2.f * xy + y2b2) / den;
        const float cb  = (1.f - x2) / den;
#pragma unroll
        for (int j = 0; j < 4; ++j) mv[j] = ca * mv[j] + cb * bv2[j];
        proj_row<4>(mv);

        floatx4 o;
#pragma unroll
        for (int j = 0; j < 4; ++j) o[j] = mv[j];
        *(floatx4*)(&out[(size_t)row * 128 + lane * 4]) = o;   // coalesced b128 store
    }
}

extern "C" void kernel_launch(void* const* d_in, const int* in_sizes, int n_in,
                              void* d_out, int out_size, void* d_ws, size_t ws_size,
                              hipStream_t stream) {
    const float* x  = (const float*)d_in[0];
    const float* W1 = (const float*)d_in[1];
    const float* b1 = (const float*)d_in[2];
    const float* W2 = (const float*)d_in[3];
    const float* b2 = (const float*)d_in[4];
    float* out = (float*)d_out;

    __bf16* w1b = (__bf16*)d_ws;               // 512*256 bf16
    __bf16* w2b = w1b + 512 * 256;             // 128*512 bf16  (total 384 KB of ws)

    const int B = in_sizes[0] / 256;           // rows (65536)

    hipLaunchKernelGGL(hypnn_cvt_w, dim3(512), dim3(256), 0, stream, W1, W2, w1b, w2b);
    hipLaunchKernelGGL(hypnn_fused, dim3(B / 64), dim3(128), 0, stream,
                       x, w1b, b1, w2b, b2, out);
}
